// TransformerBlock_51221779972854
// MI455X (gfx1250) — matrix operand order repaired and checked by reading, not compiled, not run on hardware
//
#include <hip/hip_runtime.h>
#include <hip/hip_bf16.h>

typedef _Float16 f16;
typedef __attribute__((ext_vector_type(16))) _Float16 v16h;
typedef __attribute__((ext_vector_type(8)))  float    v8f;

#define DMODEL 1024
#define SEQ    2048
#define NB     4
#define NHEAD  16
#define DK     64
#define DFF    4096
#define MROWS  (NB * SEQ)   // 8192

struct U32x8 { uint4 a, b; };

static __device__ __forceinline__ v16h make_frag(uint4 a, uint4 b) {
    U32x8 t{a, b};
    return __builtin_bit_cast(v16h, t);
}

static __device__ __forceinline__ v16h frag_ld(const f16* p0, const f16* p1) {
    uint4 a = *reinterpret_cast<const uint4*>(p0);
    uint4 b = *reinterpret_cast<const uint4*>(p1);
    return make_frag(a, b);
}

static __device__ __forceinline__ v8f vzero() {
    v8f v;
#pragma unroll
    for (int i = 0; i < 8; ++i) v[i] = 0.0f;
    return v;
}

static __device__ __forceinline__ v16h vsplat16(_Float16 s) {
    v16h v;
#pragma unroll
    for (int i = 0; i < 16; ++i) v[i] = s;
    return v;
}

static __device__ __forceinline__ v16h vscale16(v16h v, _Float16 s) {
#pragma unroll
    for (int i = 0; i < 16; ++i) v[i] = v[i] * s;
    return v;
}

static __device__ __forceinline__ v8f wmma16(v16h a, v16h b, v8f c) {
    return __builtin_amdgcn_wmma_f32_16x16x32_f16(false, a, false, b, (short)0, c, false, false);
}

// Async global -> LDS copy of 16 bytes per lane (CDNA5, tracked by ASYNCcnt).
// lds_off = low 32 bits of the generic shared-memory address (ISA 10.2: LDS
// generic addresses truncate to the in-wave LDS byte offset).
static __device__ __forceinline__ void async_cp_b128(const f16* gptr, const f16* lptr) {
    unsigned lds_off = (unsigned)(uintptr_t)lptr;
    asm volatile("global_load_async_to_lds_b128 %0, %1, off"
                 :: "v"(lds_off), "v"(gptr) : "memory");
}
static __device__ __forceinline__ void wait_async0() {
    asm volatile("s_wait_asynccnt 0x0" ::: "memory");
}

// -------------------- weight convert + transpose: W[K][N] f32 -> WT[N][K] f16
__global__ __launch_bounds__(256) void convtr_kernel(const float* __restrict__ W,
                                                     f16* __restrict__ WT,
                                                     int K, int N) {
    size_t id = (size_t)blockIdx.x * 256 + threadIdx.x;
    size_t total = (size_t)K * N;
    if (id >= total) return;
    int n = (int)(id / K);
    int k = (int)(id % K);
    WT[id] = (f16)W[(size_t)k * N + n];
}

// -------------------- layernorm row of 1024 -> f16
__global__ __launch_bounds__(256) void ln_kernel(const float* __restrict__ x,
                                                 const float* __restrict__ g,
                                                 const float* __restrict__ bb,
                                                 f16* __restrict__ out) {
    int row = blockIdx.x;
    int tid = threadIdx.x;
    const float4 xv = reinterpret_cast<const float4*>(x + (size_t)row * DMODEL)[tid];
    float vv[4] = {xv.x, xv.y, xv.z, xv.w};
    float s = vv[0] + vv[1] + vv[2] + vv[3];
    float q = vv[0]*vv[0] + vv[1]*vv[1] + vv[2]*vv[2] + vv[3]*vv[3];
#pragma unroll
    for (int m = 16; m >= 1; m >>= 1) {
        s += __shfl_xor(s, m);
        q += __shfl_xor(q, m);
    }
    __shared__ float ss[8], sq[8];
    if ((tid & 31) == 0) { ss[tid >> 5] = s; sq[tid >> 5] = q; }
    __syncthreads();
    float ts = 0.f, tq = 0.f;
#pragma unroll
    for (int i = 0; i < 8; ++i) { ts += ss[i]; tq += sq[i]; }
    float mu  = ts * (1.0f / DMODEL);
    float var = tq * (1.0f / DMODEL) - mu * mu;
    float rs  = rsqrtf(var + 1e-5f);
    int base = tid * 4;
#pragma unroll
    for (int i = 0; i < 4; ++i) {
        float y = (vv[i] - mu) * rs * g[base + i] + bb[base + i];
        out[(size_t)row * DMODEL + base + i] = (f16)y;
    }
}

// -------------------- WMMA GEMM: C[M,N] = A[M,K] * BT[N,K]^T + bias, epilogues
// BK = 64; ping-pong LDS buffers filled by async global->LDS copies.
// MODE 0: f16 out row-major
// MODE 1: f16 out with exact-erf GELU
// MODE 2: f32 out = resid + acc + bias
// MODE 3: f16 out scattered to V^T layout [B][H][DK][SEQ]
#define LDS_STRIDE 72
template <int MODE>
__global__ __launch_bounds__(256) void gemm_kernel(const f16* __restrict__ A,
                                                   const f16* __restrict__ BT,
                                                   const float* __restrict__ bias,
                                                   const float* __restrict__ resid,
                                                   void* __restrict__ outp,
                                                   int M, int N, int K) {
    __shared__ f16 As[2][128 * LDS_STRIDE];
    __shared__ f16 Bs[2][128 * LDS_STRIDE];

    const int bcol = blockIdx.x * 128;
    const int brow = blockIdx.y * 128;
    const int tid  = threadIdx.x;
    const int lane = tid & 31;
    const int wave = tid >> 5;
    const int half = lane >> 4;
    const int l16  = lane & 15;
    const int wm   = wave & 3;   // 4 waves along M
    const int wn   = wave >> 2;  // 2 waves along N

    v8f acc[2][4];
#pragma unroll
    for (int i = 0; i < 2; ++i)
#pragma unroll
        for (int j = 0; j < 4; ++j) acc[i][j] = vzero();

    const int ak = half * 16;   // A-frag k offsets within 32-chunk: {0,8}/{16,24}
    const int bk = half * 16;   // B-frag k offsets within 32-chunk: {0,8}/{16,24}

    // per-thread tile-copy coordinates (4 x 16B per matrix per K-step of 64)
    int lr[4], lk[4];
#pragma unroll
    for (int i = 0; i < 4; ++i) {
        int c2 = tid + 256 * i;
        lr[i] = c2 >> 3;         // 0..127
        lk[i] = (c2 & 7) * 8;    // 0..56
    }

    auto async_tile = [&](int buf, int k0) {
#pragma unroll
        for (int i = 0; i < 4; ++i) {
            async_cp_b128(&A[(size_t)(brow + lr[i]) * K + k0 + lk[i]],
                          &As[buf][lr[i] * LDS_STRIDE + lk[i]]);
            async_cp_b128(&BT[(size_t)(bcol + lr[i]) * K + k0 + lk[i]],
                          &Bs[buf][lr[i] * LDS_STRIDE + lk[i]]);
        }
    };

    async_tile(0, 0);
    wait_async0();
    __syncthreads();

    int buf = 0;
    for (int k0 = 0;;) {
        const bool more = (k0 + 64 < K);
        if (more) async_tile(buf ^ 1, k0 + 64);

#pragma unroll
        for (int kk = 0; kk < 2; ++kk) {
            const int kc = kk * 32;
            v16h af[2], bf[4];
#pragma unroll
            for (int mt = 0; mt < 2; ++mt) {
                int r = wm * 32 + mt * 16 + l16;
                af[mt] = frag_ld(&As[buf][r * LDS_STRIDE + kc + ak],
                                 &As[buf][r * LDS_STRIDE + kc + ak + 8]);
            }
#pragma unroll
            for (int nt = 0; nt < 4; ++nt) {
                int cn = wn * 64 + nt * 16 + l16;
                bf[nt] = frag_ld(&Bs[buf][cn * LDS_STRIDE + kc + bk],
                                 &Bs[buf][cn * LDS_STRIDE + kc + bk + 8]);
            }
#pragma unroll
            for (int mt = 0; mt < 2; ++mt)
#pragma unroll
                for (int nt = 0; nt < 4; ++nt)
                    acc[mt][nt] = wmma16(af[mt], bf[nt], acc[mt][nt]);
        }

        k0 += 64;
        if (!more) break;
        wait_async0();     // this wave's copies into buf^1 are done
        __syncthreads();   // and everyone else's too
        buf ^= 1;
    }

    // epilogue
#pragma unroll
    for (int mt = 0; mt < 2; ++mt) {
#pragma unroll
        for (int nt = 0; nt < 4; ++nt) {
            int col = bcol + wn * 64 + nt * 16 + l16;
            float bv = bias[col];
            int rowbase = brow + wm * 32 + mt * 16 + half * 8;
#pragma unroll
            for (int r = 0; r < 8; ++r) {
                int row = rowbase + r;
                float val = acc[mt][nt][r] + bv;
                if (MODE == 0) {
                    ((f16*)outp)[(size_t)row * N + col] = (f16)val;
                } else if (MODE == 1) {
                    float ge = 0.5f * val * (1.0f + erff(val * 0.70710678118654752f));
                    ((f16*)outp)[(size_t)row * N + col] = (f16)ge;
                } else if (MODE == 2) {
                    size_t idx = (size_t)row * N + col;
                    ((float*)outp)[idx] = resid[idx] + val;
                } else {  // MODE 3: V^T [B][H][DK][SEQ]
                    int b = row >> 11, s = row & 2047;
                    int h = col >> 6,  d = col & 63;
                    ((f16*)outp)[((size_t)((b << 4) + h) * DK + d) * SEQ + s] = (f16)val;
                }
            }
        }
    }
}

// -------------------- flash attention: per block = 128 q rows of one (b,h)
// 64 keys per iteration; row sums via P @ ones WMMA (no shuffle sum-reduction).
__global__ __launch_bounds__(256) void attn_kernel(const f16* __restrict__ q16,
                                                   const f16* __restrict__ k16,
                                                   const f16* __restrict__ vt,
                                                   f16* __restrict__ ctx16) {
    __shared__ f16 Pbuf[8][16][64];   // 16 KB, per-wave private slice

    const int qb = blockIdx.x & 15;           // SEQ/128 = 16
    const int h  = (blockIdx.x >> 4) & 15;
    const int b  = blockIdx.x >> 8;
    const int lane = threadIdx.x & 31;
    const int wave = threadIdx.x >> 5;
    const int half = lane >> 4;
    const int l16  = lane & 15;
    const int qrow0 = qb * 128 + wave * 16;

    // Q A-fragments (16 rows x 64 dk -> two k=32 chunks); fold in 1/sqrt(dk)
    v16h qa[2];
    {
        int qr = qrow0 + l16;
        const f16* qp = q16 + ((size_t)(b * SEQ + qr)) * DMODEL + h * DK;
#pragma unroll
        for (int kk = 0; kk < 2; ++kk) {
            const f16* p = qp + kk * 32 + half * 16;
            qa[kk] = vscale16(frag_ld(p, p + 8), (_Float16)0.125f);
        }
    }
    const v16h ones = vsplat16((_Float16)1.0f);

    v8f o[4], osum;
#pragma unroll
    for (int t = 0; t < 4; ++t) o[t] = vzero();
    osum = vzero();
    float mrun[8];
#pragma unroll
    for (int r = 0; r < 8; ++r) mrun[r] = -1e30f;

    for (int j0 = 0; j0 < SEQ; j0 += 64) {
        // ---- scores for 64 keys: four 16-col tiles, K-dim = dk (2 chunks)
        v8f sc[4];
#pragma unroll
        for (int n = 0; n < 4; ++n) sc[n] = vzero();
#pragma unroll
        for (int n = 0; n < 4; ++n) {
            int key = j0 + n * 16 + l16;
            const f16* kp = k16 + ((size_t)(b * SEQ + key)) * DMODEL + h * DK;
#pragma unroll
            for (int kk = 0; kk < 2; ++kk) {
                const f16* p = kp + kk * 32 + half * 16;
                v16h kb = frag_ld(p, p + 8);
                sc[n] = wmma16(qa[kk], kb, sc[n]);
            }
        }

        // ---- online softmax over the 64 new columns (max via 16-lane shuffles,
        //      sums handled later by the ones-matrix WMMA)
        const int prow = (half ? 8 : 0);
#pragma unroll
        for (int r = 0; r < 8; ++r) {
            float mx = fmaxf(fmaxf(sc[0][r], sc[1][r]), fmaxf(sc[2][r], sc[3][r]));
#pragma unroll
            for (int m = 8; m >= 1; m >>= 1) mx = fmaxf(mx, __shfl_xor(mx, m));
            float newm = fmaxf(mrun[r], mx);
            float corr = __expf(mrun[r] - newm);
            mrun[r] = newm;
#pragma unroll
            for (int t = 0; t < 4; ++t) o[t][r] *= corr;
            osum[r] *= corr;
#pragma unroll
            for (int n = 0; n < 4; ++n) {
                float p = __expf(sc[n][r] - newm);
                Pbuf[wave][prow + r][n * 16 + l16] = (f16)p;
            }
        }

        // ---- P (16x64) A-fragments from per-wave LDS
        v16h pa[2];
#pragma unroll
        for (int kk = 0; kk < 2; ++kk) {
            const f16* pp = &Pbuf[wave][l16][kk * 32 + half * 16];
            pa[kk] = frag_ld(pp, pp + 8);
        }

        // ---- row sums: osum += P @ ones
        osum = wmma16(pa[0], ones, osum);
        osum = wmma16(pa[1], ones, osum);

        // ---- o += P @ V  (V^T rows contiguous in s)
#pragma unroll
        for (int t = 0; t < 4; ++t) {
            int d = t * 16 + l16;
            const f16* vp0 = vt + ((size_t)((b << 4) + h) * DK + d) * SEQ + j0;
#pragma unroll
            for (int kk = 0; kk < 2; ++kk) {
                const f16* p = vp0 + kk * 32 + half * 16;
                v16h vb = frag_ld(p, p + 8);
                o[t] = wmma16(pa[kk], vb, o[t]);
            }
        }
    }

    // ---- normalize and write ctx in [B,S,D] f16 layout
#pragma unroll
    for (int r = 0; r < 8; ++r) {
        float inv = 1.0f / osum[r];
        int row = qrow0 + (half ? 8 : 0) + r;
#pragma unroll
        for (int t = 0; t < 4; ++t) {
            float val = o[t][r] * inv;
            ctx16[((size_t)(b * SEQ + row)) * DMODEL + h * DK + t * 16 + l16] = (f16)val;
        }
    }
}

extern "C" void kernel_launch(void* const* d_in, const int* in_sizes, int n_in,
                              void* d_out, int out_size, void* d_ws, size_t ws_size,
                              hipStream_t stream) {
    const float* x     = (const float*)d_in[0];
    const float* Wq    = (const float*)d_in[1];
    const float* bq    = (const float*)d_in[2];
    const float* Wk    = (const float*)d_in[3];
    const float* bk    = (const float*)d_in[4];
    const float* Wv    = (const float*)d_in[5];
    const float* bv    = (const float*)d_in[6];
    const float* Wo    = (const float*)d_in[7];
    const float* bo    = (const float*)d_in[8];
    const float* ln1_g = (const float*)d_in[9];
    const float* ln1_b = (const float*)d_in[10];
    const float* W1    = (const float*)d_in[11];
    const float* b1    = (const float*)d_in[12];
    const float* W2    = (const float*)d_in[13];
    const float* b2    = (const float*)d_in[14];
    const float* ln2_g = (const float*)d_in[15];
    const float* ln2_b = (const float*)d_in[16];
    float* out = (float*)d_out;

    const size_t MB = 1u << 20;
    char* ws = (char*)d_ws;
    f16* WqT  = (f16*)(ws + 0 * MB);    // 2 MB
    f16* WkT  = (f16*)(ws + 2 * MB);    // 2 MB
    f16* WvT  = (f16*)(ws + 4 * MB);    // 2 MB
    f16* WoT  = (f16*)(ws + 6 * MB);    // 2 MB
    f16* W1T  = (f16*)(ws + 8 * MB);    // 8 MB
    f16* W2T  = (f16*)(ws + 16 * MB);   // 8 MB
    f16* h16  = (f16*)(ws + 24 * MB);   // 16 MB
    f16* q16  = (f16*)(ws + 40 * MB);   // 16 MB
    f16* k16  = (f16*)(ws + 56 * MB);   // 16 MB
    f16* vt   = (f16*)(ws + 72 * MB);   // 16 MB  [B][H][DK][SEQ]
    f16* c16  = (f16*)(ws + 88 * MB);   // 16 MB
    f16* h2   = (f16*)(ws + 104 * MB);  // 16 MB
    f16* ff16 = (f16*)(ws + 120 * MB);  // 64 MB

    // weight convert/transpose
    convtr_kernel<<<(DMODEL * DMODEL) / 256, 256, 0, stream>>>(Wq, WqT, DMODEL, DMODEL);
    convtr_kernel<<<(DMODEL * DMODEL) / 256, 256, 0, stream>>>(Wk, WkT, DMODEL, DMODEL);
    convtr_kernel<<<(DMODEL * DMODEL) / 256, 256, 0, stream>>>(Wv, WvT, DMODEL, DMODEL);
    convtr_kernel<<<(DMODEL * DMODEL) / 256, 256, 0, stream>>>(Wo, WoT, DMODEL, DMODEL);
    convtr_kernel<<<(DMODEL * DFF) / 256, 256, 0, stream>>>(W1, W1T, DMODEL, DFF);
    convtr_kernel<<<(DFF * DMODEL) / 256, 256, 0, stream>>>(W2, W2T, DFF, DMODEL);

    // LN1
    ln_kernel<<<MROWS, 256, 0, stream>>>(x, ln1_g, ln1_b, h16);

    // Q, K, V projections
    dim3 gQ(DMODEL / 128, MROWS / 128);
    gemm_kernel<0><<<gQ, 256, 0, stream>>>(h16, WqT, bq, nullptr, (void*)q16, MROWS, DMODEL, DMODEL);
    gemm_kernel<0><<<gQ, 256, 0, stream>>>(h16, WkT, bk, nullptr, (void*)k16, MROWS, DMODEL, DMODEL);
    gemm_kernel<3><<<gQ, 256, 0, stream>>>(h16, WvT, bv, nullptr, (void*)vt,  MROWS, DMODEL, DMODEL);

    // attention
    attn_kernel<<<NB * NHEAD * (SEQ / 128), 256, 0, stream>>>(q16, k16, vt, c16);

    // output projection + residual into d_out
    gemm_kernel<2><<<gQ, 256, 0, stream>>>(c16, WoT, bo, x, (void*)out, MROWS, DMODEL, DMODEL);

    // LN2
    ln_kernel<<<MROWS, 256, 0, stream>>>(out, ln2_g, ln2_b, h2);

    // FFN
    dim3 gF1(DFF / 128, MROWS / 128);
    gemm_kernel<1><<<gF1, 256, 0, stream>>>(h2, W1T, b1, nullptr, (void*)ff16, MROWS, DFF, DMODEL);
    gemm_kernel<2><<<gQ, 256, 0, stream>>>(ff16, W2T, b2, out, (void*)out, MROWS, DMODEL, DFF);
}
